// YuaAttention_61881888800979
// MI455X (gfx1250) — matrix operand order repaired and checked by reading, not compiled, not run on hardware
//
#include <hip/hip_runtime.h>
#include <hip/hip_bf16.h>
#include <cstdint>

typedef __attribute__((ext_vector_type(16))) __bf16 v16bf;
typedef __attribute__((ext_vector_type(8)))  __bf16 v8bf;
typedef __attribute__((ext_vector_type(4)))  __bf16 v4bf;
typedef __attribute__((ext_vector_type(8)))  float  v8f;
typedef __attribute__((ext_vector_type(4)))  int    v4i;
typedef __attribute__((ext_vector_type(4)))  unsigned int v4u;
typedef __attribute__((ext_vector_type(8)))  int    v8i;

#define NUM_HEADS   16
#define NUM_KV      4
#define HD          128
#define BATCH       2
#define SEQ         2048
#define HID         2048
#define NTOK        (BATCH * SEQ)
#define KVW         (NUM_KV * HD)   // 512

#define USE_TDM 1   // Tensor Data Mover staging (inline asm, toolchain-portable)

// ---------------- CDNA5 async global->LDS copy (ASYNCcnt path) -------------
#if defined(__has_builtin)
# if __has_builtin(__builtin_amdgcn_global_load_async_to_lds_b128)
#  define USE_ASYNC 1
# endif
#endif
#ifndef USE_ASYNC
# define USE_ASYNC 0
#endif

typedef __attribute__((address_space(1))) v4i as1_v4i;  // global int4
typedef __attribute__((address_space(3))) v4i as3_v4i;  // LDS int4

__device__ __forceinline__ void async_b128(const void* g, void* l) {
#if USE_ASYNC
  __builtin_amdgcn_global_load_async_to_lds_b128(
      (as1_v4i*)(uintptr_t)g, (as3_v4i*)(uint32_t)(uintptr_t)l, 0, 0);
#else
  *(v8bf*)l = *(const v8bf*)g;
#endif
}

#if USE_ASYNC
# define ASYNC_WAIT(n) asm volatile("s_wait_asynccnt " #n ::: "memory")
#else
# define ASYNC_WAIT(n)
#endif

// ---------------- CDNA5 Tensor Data Mover: 2-D tile -> LDS -----------------
// D# per ISA 8.3/8.4: group0 = {count, lds_addr, global_addr, type=2},
// group1 = {data_size=2B, tensor dims, tile dims, 48-bit dim0 stride}.
// 2-D form: groups 2/3 omitted. Issued by one wave; tracked with TENSORcnt.
__device__ __forceinline__ void tdm_load_2d(const void* gaddr, void* laddr,
                                            int tile0, int tile1,
                                            unsigned long long stride0_elems) {
  const uint64_t ga = (uint64_t)(uintptr_t)gaddr;        // byte address
  const uint32_t la = (uint32_t)(uintptr_t)laddr;        // LDS byte offset
  v4u g0;
  g0[0] = 1u;                                            // count=1, user-mode
  g0[1] = la;                                            // lds_addr
  g0[2] = (uint32_t)ga;                                  // global_addr[31:0]
  g0[3] = (uint32_t)((ga >> 32) & 0x01ffffffu) | 0x80000000u;  // [56:32]|type=2
  v8i g1;
  const unsigned d0 = (unsigned)tile0;                   // tensor_dim0 == tile0
  const unsigned d1 = (unsigned)tile1;                   // tensor_dim1 == tile1
  const uint64_t s0 = stride0_elems;                     // in data_size units
  g1[0] = (int)(1u << 16);                               // data_size=1 (2B)
  g1[1] = (int)((d0 & 0xffffu) << 16);                   // tensor_dim0[15:0]
  g1[2] = (int)(((d0 >> 16) & 0xffffu) | ((d1 & 0xffffu) << 16));
  g1[3] = (int)(((d1 >> 16) & 0xffffu) | (((unsigned)tile0 & 0xffffu) << 16));
  g1[4] = (int)((unsigned)tile1 & 0xffffu);              // tile_dim1; tile_dim2=0
  g1[5] = (int)(uint32_t)s0;                             // dim0_stride[31:0]
  g1[6] = (int)((uint32_t)(s0 >> 32) & 0xffffu);         // dim0_stride[47:32]
  g1[7] = 0;
  asm volatile("tensor_load_to_lds %0, %1" :: "s"(g0), "s"(g1) : "memory");
}

#if USE_TDM
# define TILE_WAIT(n) asm volatile("s_wait_tensorcnt " #n ::: "memory")
#else
# define TILE_WAIT(n) ASYNC_WAIT(n)
#endif

// ---------------------------------------------------------------------------
// fp32 -> bf16 bulk convert (vec4)
// ---------------------------------------------------------------------------
__global__ __launch_bounds__(256)
void f32_to_bf16_kernel(const float* __restrict__ in, __bf16* __restrict__ out,
                        int n4) {
  int i = blockIdx.x * 256 + threadIdx.x;
  if (i >= n4) return;
  float4 v = ((const float4*)in)[i];
  v4bf o;
  o[0] = (__bf16)v.x; o[1] = (__bf16)v.y; o[2] = (__bf16)v.z; o[3] = (__bf16)v.w;
  ((v4bf*)out)[i] = o;
}

// ---------------------------------------------------------------------------
// GEMM: C[M,N] = A[M,K] * B[N,K]^T, A/B bf16 K-contig, C fp32 or bf16.
// Block tile 128x128, 8 waves as 4(M) x 2(N); wave tile 32x64 = 2x4 WMMA frags.
// Double-buffered LDS tiles staged by the Tensor Data Mover (wave 0 issues a
// descriptor per tile); copy of tile t+1 overlaps WMMA compute of tile t.
// ---------------------------------------------------------------------------
#define TM 128
#define TN 128
#define TK 32

__global__ __launch_bounds__(256)
void gemm_nt_kernel(const __bf16* __restrict__ A, const __bf16* __restrict__ B,
                    void* __restrict__ Cv, int M, int N, int K, int c_bf16) {
  __shared__ alignas(32) __bf16 lA[2][TM * TK];   // 2 x 8 KB
  __shared__ alignas(32) __bf16 lB[2][TN * TK];   // 2 x 8 KB

  const int tid  = threadIdx.x;
  const int lane = tid & 31;
  const int wave = tid >> 5;
  const int hl   = lane >> 4;
  const int l16  = lane & 15;

  const int bm = blockIdx.y * TM;
  const int bn = blockIdx.x * TN;
  const int wm = (wave & 3) * 32;
  const int wn = (wave >> 2) * 64;

  // stage one 128x32 A/B tile pair
  auto stage = [&](int buf, int k0) {
#if USE_TDM
    if (wave == 0) {   // 2 TDM descriptors move both 8 KB tiles
      tdm_load_2d(&A[(size_t)bm * K + k0], &lA[buf][0], TK, TM, (unsigned long long)K);
      tdm_load_2d(&B[(size_t)bn * K + k0], &lB[buf][0], TK, TN, (unsigned long long)K);
    }
#else
#pragma unroll
    for (int t2 = 0; t2 < 2; ++t2) {
      int idx = tid + t2 * 256;
      int r = idx >> 2, c8 = (idx & 3) * 8;
      async_b128(&A[(size_t)(bm + r) * K + k0 + c8], &lA[buf][idx * 8]);
      async_b128(&B[(size_t)(bn + r) * K + k0 + c8], &lB[buf][idx * 8]);
    }
#endif
  };

  v8f acc[2][4] = {};
  const int nk = K / TK;

  stage(0, 0);
  for (int t = 0; t < nk; ++t) {
    const int buf = t & 1;
    if (t + 1 < nk) {
      stage(1 - buf, (t + 1) * TK);   // overlap with compute of tile t
      if (t + 2 < nk) {               // prefetch tile t+2 (global_prefetch_b8)
        __builtin_prefetch(&A[(size_t)(bm + (tid >> 1)) * K + (t + 2) * TK], 0, 3);
        __builtin_prefetch(&B[(size_t)(bn + (tid >> 1)) * K + (t + 2) * TK], 0, 3);
      }
#if USE_TDM
      TILE_WAIT(2);                   // wave0: tile t done (in-order); others 0
#else
      TILE_WAIT(4);
#endif
    } else {
      TILE_WAIT(0);
    }
    __syncthreads();

    v16bf af[2], bfr[4];
#pragma unroll
    for (int i = 0; i < 2; ++i) {
      const __bf16* base = &lA[buf][(wm + i * 16 + l16) * TK + hl * 8];
      v8bf lo = *(const v8bf*)base;
      v8bf hi = *(const v8bf*)(base + 16);
#pragma unroll
      for (int e = 0; e < 8; ++e) { af[i][e] = lo[e]; af[i][e + 8] = hi[e]; }
    }
#pragma unroll
    for (int j = 0; j < 4; ++j)
    {
      const __bf16* bb = &lB[buf][(wn + j * 16 + l16) * TK + hl * 8];
      v8bf lo = *(const v8bf*)bb;
      v8bf hi = *(const v8bf*)(bb + 16);
#pragma unroll
      for (int e = 0; e < 8; ++e) { bfr[j][e] = lo[e]; bfr[j][e + 8] = hi[e]; }
    }

#pragma unroll
    for (int i = 0; i < 2; ++i)
#pragma unroll
      for (int j = 0; j < 4; ++j)
        acc[i][j] = __builtin_amdgcn_wmma_f32_16x16x32_bf16(
            false, af[i], false, bfr[j], (short)0, acc[i][j], false, false);
    __syncthreads();   // all waves done reading buf before it is re-staged
  }

  if (c_bf16) {
    __bf16* C = (__bf16*)Cv;
#pragma unroll
    for (int i = 0; i < 2; ++i)
#pragma unroll
      for (int j = 0; j < 4; ++j)
#pragma unroll
        for (int r = 0; r < 8; ++r) {
          int m = bm + wm + i * 16 + r + 8 * hl;
          int n = bn + wn + j * 16 + l16;
          C[(size_t)m * N + n] = (__bf16)acc[i][j][r];
        }
  } else {
    float* C = (float*)Cv;
#pragma unroll
    for (int i = 0; i < 2; ++i)
#pragma unroll
      for (int j = 0; j < 4; ++j)
#pragma unroll
        for (int r = 0; r < 8; ++r) {
          int m = bm + wm + i * 16 + r + 8 * hl;
          int n = bn + wn + j * 16 + l16;
          C[(size_t)m * N + n] = acc[i][j][r];
        }
  }
}

// ---------------------------------------------------------------------------
// Fused QK-RMSNorm + RoPE on bf16 in place (norm math in fp32).
// ---------------------------------------------------------------------------
__global__ __launch_bounds__(256)
void rmsnorm_rope_kernel(__bf16* __restrict__ X, const float* __restrict__ w,
                         int n_heads) {
  const int gwave = (blockIdx.x * blockDim.x + threadIdx.x) >> 5;
  const int lane  = threadIdx.x & 31;
  if (gwave >= NTOK * n_heads) return;

  const int token = gwave / n_heads;
  const int head  = gwave % n_heads;
  const int pos   = token % SEQ;

  __bf16* x = X + (size_t)token * n_heads * HD + head * HD;

  float v[4];
  float ss = 0.f;
#pragma unroll
  for (int i = 0; i < 4; ++i) { v[i] = (float)x[lane * 4 + i]; ss += v[i] * v[i]; }
#pragma unroll
  for (int off = 16; off >= 1; off >>= 1) ss += __shfl_xor(ss, off, 32);
  const float rinv = rsqrtf(ss * (1.0f / HD) + 1e-5f);

  float out[4];
#pragma unroll
  for (int i = 0; i < 4; ++i) {
    int d = lane * 4 + i;
    float val = v[i] * rinv * w[d];
    float partner = __shfl_xor(val, 16, 32);   // dim d ^ 64
    float ang = (float)pos * __powf(500000.0f, -(float)(d & 63) * (1.0f / 64.0f));
    float c = __cosf(ang), s = __sinf(ang);
    float rh = (d < 64) ? -partner : partner;
    out[i] = val * c + rh * s;
  }
#pragma unroll
  for (int i = 0; i < 4; ++i) x[lane * 4 + i] = (__bf16)out[i];
}

// ---------------------------------------------------------------------------
// Causal flash attention, GQA, bf16 operands, fp32 online softmax.
// Block = (b, h, 128 q-rows); wave owns 16 q-rows. Q and K tiles staged by the
// TDM; double-buffered K/V tiles alias the dead-after-prologue Q region.
// ---------------------------------------------------------------------------
#define QT 128
#define KB 32

__global__ __launch_bounds__(256)
void flash_attn_kernel(const __bf16* __restrict__ Q, const __bf16* __restrict__ Kv,
                       const __bf16* __restrict__ Vv, __bf16* __restrict__ O) {
  const int bh  = blockIdx.y;
  const int b   = bh / NUM_HEADS;
  const int h   = bh % NUM_HEADS;
  const int kvh = h / (NUM_HEADS / NUM_KV);
  const int q0  = blockIdx.x * QT;

  const int tid  = threadIdx.x;
  const int lane = tid & 31;
  const int wave = tid >> 5;
  const int hl   = lane >> 4;
  const int l16  = lane & 15;

  __shared__ alignas(32) unsigned char smem[40960];
  __bf16* lQ   = (__bf16*)smem;              // 32 KB (dead after prologue)
  __bf16* lKb  = (__bf16*)smem;              // 2 x 8 KB (aliases lQ)
  __bf16* lVtb = (__bf16*)(smem + 16384);    // 2 x 8 KB (aliases lQ)
  __bf16* lP   = (__bf16*)(smem + 32768) + wave * 16 * KB;  // 8 x 1 KB

  // stage Q tile (128 x 128, row stride 2048)
  const __bf16* Qbase = Q + (size_t)(b * SEQ + q0) * (NUM_HEADS * HD) + h * HD;
#if USE_TDM
  if (wave == 0) tdm_load_2d(Qbase, lQ, HD, QT, NUM_HEADS * HD);
#else
#pragma unroll
  for (int t = 0; t < 8; ++t) {
    int idx = tid + t * 256;
    int r = idx >> 4, c8 = (idx & 15) * 8;
    async_b128(&Qbase[(size_t)r * (NUM_HEADS * HD) + c8], &lQ[idx * 8]);
  }
#endif
  TILE_WAIT(0);
  __syncthreads();

  v16bf qa[4];
#pragma unroll
  for (int kk = 0; kk < 4; ++kk) {
    const __bf16* base = &lQ[(wave * 16 + l16) * HD + kk * 32 + hl * 8];
    v8bf lo = *(const v8bf*)base;
    v8bf hi = *(const v8bf*)(base + 16);
#pragma unroll
    for (int e = 0; e < 8; ++e) { qa[kk][e] = lo[e]; qa[kk][e + 8] = hi[e]; }
  }
  __syncthreads();   // all waves finished with lQ; K/V buffers may overwrite it

  v8f oacc[8] = {};
  float rmax[8], rsum[8];
#pragma unroll
  for (int r = 0; r < 8; ++r) { rmax[r] = -3.0e38f; rsum[r] = 0.f; }
  const float scale = 0.08838834764831845f;  // 1/sqrt(128)

  const __bf16* Kbase = Kv + (size_t)(b * SEQ) * KVW + kvh * HD;
  const __bf16* Vbase = Vv + (size_t)(b * SEQ) * KVW + kvh * HD;

  // stage one 32-key K/V tile: K via TDM, V transposed manually into LDS
  auto stage_kv = [&](int buf, int j0) {
    __bf16* lK  = lKb  + buf * (KB * HD);
    __bf16* lVt = lVtb + buf * (HD * KB);
#if USE_TDM
    if (wave == 0) tdm_load_2d(&Kbase[(size_t)j0 * KVW], lK, HD, KB, KVW);
#endif
#pragma unroll
    for (int t = 0; t < 2; ++t) {
      int idx = tid + t * 256;
      int r = idx >> 4, c8 = (idx & 15) * 8;
#if !USE_TDM
      async_b128(&Kbase[(size_t)(j0 + r) * KVW + c8], &lK[idx * 8]);
#endif
      v8bf vv = *(const v8bf*)&Vbase[(size_t)(j0 + r) * KVW + c8];
#pragma unroll
      for (int e = 0; e < 8; ++e) lVt[(c8 + e) * KB + r] = vv[e];
    }
  };

  const int ntile = (q0 + QT) / KB;  // causal bound
  stage_kv(0, 0);
  for (int jt = 0; jt < ntile; ++jt) {
    const int buf = jt & 1;
    const int j0  = jt * KB;
    if (jt + 1 < ntile) {
      stage_kv(1 - buf, j0 + KB);   // overlap with compute of tile jt
#if USE_TDM
      TILE_WAIT(1);
#else
      TILE_WAIT(2);
#endif
    } else {
      TILE_WAIT(0);
    }
    __syncthreads();

    const __bf16* lK  = lKb  + buf * (KB * HD);
    const __bf16* lVt = lVtb + buf * (HD * KB);

    // S = Q K^T
    v8f sf[2] = {};
#pragma unroll
    for (int g = 0; g < 2; ++g)
#pragma unroll
      for (int kk = 0; kk < 4; ++kk) {
        v16bf bfrag;
        {
          const __bf16* bb = &lK[(g * 16 + l16) * HD + kk * 32 + hl * 8];
          v8bf lo = *(const v8bf*)bb;
          v8bf hi = *(const v8bf*)(bb + 16);
#pragma unroll
          for (int e = 0; e < 8; ++e) { bfrag[e] = lo[e]; bfrag[e + 8] = hi[e]; }
        }
        sf[g] = __builtin_amdgcn_wmma_f32_16x16x32_bf16(
            false, qa[kk], false, bfrag, (short)0, sf[g], false, false);
      }

    // scale + causal mask + online softmax
    float pnew[2][8];
#pragma unroll
    for (int r = 0; r < 8; ++r) {
      int m = q0 + wave * 16 + r + 8 * hl;
      float rowmax = -3.0e38f;
#pragma unroll
      for (int g = 0; g < 2; ++g) {
        int key = j0 + g * 16 + l16;
        float s = sf[g][r] * scale;
        if (key > m) s = -3.0e38f;
        sf[g][r] = s;
        rowmax = fmaxf(rowmax, s);
      }
#pragma unroll
      for (int off = 8; off >= 1; off >>= 1)
        rowmax = fmaxf(rowmax, __shfl_xor(rowmax, off, 32));
      float nm   = fmaxf(rmax[r], rowmax);
      float corr = __expf(rmax[r] - nm);
      rmax[r] = nm;
      float psum = 0.f;
#pragma unroll
      for (int g = 0; g < 2; ++g) {
        float p = __expf(sf[g][r] - nm);
        pnew[g][r] = p;
        psum += p;
      }
#pragma unroll
      for (int off = 8; off >= 1; off >>= 1) psum += __shfl_xor(psum, off, 32);
      rsum[r] = rsum[r] * corr + psum;
#pragma unroll
      for (int n = 0; n < 8; ++n) oacc[n][r] *= corr;
    }

    // transpose P (C-layout -> A-layout) through per-wave LDS scratch
#pragma unroll
    for (int g = 0; g < 2; ++g)
#pragma unroll
      for (int r = 0; r < 8; ++r)
        lP[(r + 8 * hl) * KB + g * 16 + l16] = (__bf16)pnew[g][r];
    asm volatile("s_wait_dscnt 0" ::: "memory");  // same-wave LDS RAW

    v16bf pa;
    {
      const __bf16* base = &lP[l16 * KB + hl * 8];
      v8bf lo = *(const v8bf*)base;
      v8bf hi = *(const v8bf*)(base + 16);
#pragma unroll
      for (int e = 0; e < 8; ++e) { pa[e] = lo[e]; pa[e + 8] = hi[e]; }
    }

    // O += P * V
#pragma unroll
    for (int n = 0; n < 8; ++n) {
      v16bf bfrag;
      {
        const __bf16* bb = &lVt[(n * 16 + l16) * KB + hl * 8];
        v8bf lo = *(const v8bf*)bb;
        v8bf hi = *(const v8bf*)(bb + 16);
#pragma unroll
        for (int e = 0; e < 8; ++e) { bfrag[e] = lo[e]; bfrag[e + 8] = hi[e]; }
      }
      oacc[n] = __builtin_amdgcn_wmma_f32_16x16x32_bf16(
          false, pa, false, bfrag, (short)0, oacc[n], false, false);
    }
    __syncthreads();   // all waves done with buf before it is re-staged
  }

  // normalize, store bf16 attn output [token, h*128 + d]
  __bf16* Obase = O + (size_t)(b * SEQ + q0 + wave * 16) * (NUM_HEADS * HD) + h * HD;
#pragma unroll
  for (int n = 0; n < 8; ++n)
#pragma unroll
    for (int r = 0; r < 8; ++r) {
      int m = r + 8 * hl;
      Obase[(size_t)m * (NUM_HEADS * HD) + n * 16 + l16] = (__bf16)(oacc[n][r] / rsum[r]);
    }
}

// ---------------------------------------------------------------------------
extern "C" void kernel_launch(void* const* d_in, const int* in_sizes, int n_in,
                              void* d_out, int out_size, void* d_ws, size_t ws_size,
                              hipStream_t stream) {
  (void)in_sizes; (void)n_in; (void)out_size; (void)ws_size;
  const float* hidden   = (const float*)d_in[0];
  const float* q_w      = (const float*)d_in[1];
  const float* k_w      = (const float*)d_in[2];
  const float* v_w      = (const float*)d_in[3];
  const float* o_w      = (const float*)d_in[4];
  const float* q_norm_w = (const float*)d_in[5];
  const float* k_norm_w = (const float*)d_in[6];
  float* out = (float*)d_out;

  // bf16 workspace layout
  __bf16* hb  = (__bf16*)d_ws;                       // [4096,2048]
  __bf16* qwb = hb  + (size_t)NTOK * HID;            // [2048,2048]
  __bf16* kwb = qwb + (size_t)HID * HID;             // [512,2048]
  __bf16* vwb = kwb + (size_t)KVW * HID;             // [512,2048]
  __bf16* owb = vwb + (size_t)KVW * HID;             // [2048,2048]
  __bf16* Qb  = owb + (size_t)HID * HID;             // [4096,2048]
  __bf16* Kb  = Qb  + (size_t)NTOK * HID;            // [4096,512]
  __bf16* Vb  = Kb  + (size_t)NTOK * KVW;            // [4096,512]
  __bf16* Ab  = Vb  + (size_t)NTOK * KVW;            // [4096,2048]

  // one-time fp32 -> bf16 conversion of activations + weights
  f32_to_bf16_kernel<<<(NTOK * HID / 4 + 255) / 256, 256, 0, stream>>>(hidden, hb, NTOK * HID / 4);
  f32_to_bf16_kernel<<<(HID * HID / 4 + 255) / 256, 256, 0, stream>>>(q_w, qwb, HID * HID / 4);
  f32_to_bf16_kernel<<<(KVW * HID / 4 + 255) / 256, 256, 0, stream>>>(k_w, kwb, KVW * HID / 4);
  f32_to_bf16_kernel<<<(KVW * HID / 4 + 255) / 256, 256, 0, stream>>>(v_w, vwb, KVW * HID / 4);
  f32_to_bf16_kernel<<<(HID * HID / 4 + 255) / 256, 256, 0, stream>>>(o_w, owb, HID * HID / 4);

  // QKV projections (bf16 outputs)
  gemm_nt_kernel<<<dim3(HID / TN, NTOK / TM), 256, 0, stream>>>(hb, qwb, Qb, NTOK, HID, HID, 1);
  gemm_nt_kernel<<<dim3(KVW / TN, NTOK / TM), 256, 0, stream>>>(hb, kwb, Kb, NTOK, KVW, HID, 1);
  gemm_nt_kernel<<<dim3(KVW / TN, NTOK / TM), 256, 0, stream>>>(hb, vwb, Vb, NTOK, KVW, HID, 1);

  // QK RMSNorm + RoPE (in place, bf16)
  rmsnorm_rope_kernel<<<(NTOK * NUM_HEADS) / 8, 256, 0, stream>>>(Qb, q_norm_w, NUM_HEADS);
  rmsnorm_rope_kernel<<<(NTOK * NUM_KV) / 8, 256, 0, stream>>>(Kb, k_norm_w, NUM_KV);

  // causal flash attention with GQA
  flash_attn_kernel<<<dim3(SEQ / QT, BATCH * NUM_HEADS), 256, 0, stream>>>(Qb, Kb, Vb, Ab);

  // output projection (fp32 out)
  gemm_nt_kernel<<<dim3(HID / TN, NTOK / TM), 256, 0, stream>>>(Ab, owb, out, NTOK, HID, HID, 0);
}
